// Gemini_78039555769096
// MI455X (gfx1250) — compile-verified, matrix operand order checked by reading, not run on hardware
//
#include <hip/hip_runtime.h>
#include <hip/hip_bf16.h>

typedef __attribute__((ext_vector_type(16))) __bf16 v16bf;
typedef __attribute__((ext_vector_type(8)))  __bf16 v8bf;
typedef __attribute__((ext_vector_type(8)))  float  v8f;

#define B_   32
#define T_   1000
#define NI_  128
#define NR_  512
#define NWG  8
#define NCOL (NR_ / NWG)            // 64 neuron columns per workgroup
#define DECAY_V 0.95122942450071400910f   // exp(-1/20)
#define OUT_DECAY 0.8f

__device__ inline unsigned short f2bf_bits(float f) {
  unsigned u = __builtin_bit_cast(unsigned, f);
  unsigned r = u + 0x7FFFu + ((u >> 16) & 1u);   // round-to-nearest-even
  return (unsigned short)(r >> 16);
}

// ---------------------------------------------------------------------------
// Init: zero the double-buffered spike staging buffer and the grid barrier.
// ---------------------------------------------------------------------------
__global__ void lsnn_init(unsigned* __restrict__ zbuf_u32, unsigned* __restrict__ bar) {
  int tid = blockIdx.x * blockDim.x + threadIdx.x;
  for (int i = tid; i < (2 * B_ * NR_) / 2; i += blockDim.x * gridDim.x)
    zbuf_u32[i] = 0u;
  if (tid == 0) { bar[0] = 0u; bar[1] = 0u; }
}

// ---------------------------------------------------------------------------
// Pre-pass: inputs [B][T][NI] f32  ->  zin [T][B][NI] bf16 (time-major).
// Bandwidth-bound; removes all f32->bf16 conversion from the recurrence loop.
// ---------------------------------------------------------------------------
__global__ void __launch_bounds__(256) lsnn_convert_inputs(
    const float* __restrict__ inputs, unsigned short* __restrict__ zin)
{
  size_t idx = (size_t)blockIdx.x * blockDim.x + threadIdx.x;   // over T*B*NI
  if (idx >= (size_t)T_ * B_ * NI_) return;
  int i  = (int)(idx % NI_);
  int tb = (int)(idx / NI_);
  int b  = tb % B_;
  int t  = tb / B_;
  zin[idx] = f2bf_bits(inputs[((size_t)b * T_ + t) * NI_ + i]);
}

// ---------------------------------------------------------------------------
// Persistent LIF recurrence. 8 WGs x 256 threads (8 waves). WG g owns neuron
// columns [g*64, g*64+64). w_rec slice (bf16, transposed, autapse-masked) and
// w_in slice live in LDS. Membrane state lives in registers in WMMA D layout.
// ---------------------------------------------------------------------------
__global__ void __launch_bounds__(256) lsnn_step_kernel(
    const __bf16* __restrict__ zin,     // [T][B][NI] bf16
    const float* __restrict__ w_in,     // [NI][NR]
    const float* __restrict__ w_rec,    // [NR][NR]
    unsigned short* __restrict__ zbuf,  // [2][B][NR] bf16 staging
    unsigned* __restrict__ bar,         // {counter, generation}
    float* __restrict__ voltages,       // [B][T][NR]
    float* __restrict__ spikes)         // [B][T][NR]
{
  extern __shared__ __bf16 smem[];
  __bf16* wrec_t = smem;                  // [NCOL][NR_]  (N-major, K contiguous)
  __bf16* win_t  = smem + NCOL * NR_;     // [NCOL][NI_]

  const int g   = blockIdx.x;
  const int tid = threadIdx.x;

  // Stage transposed bf16 weight tiles into LDS; zero the autapse diagonal.
  for (int idx = tid; idx < NCOL * NR_; idx += 256) {
    int nl2 = idx / NR_, k = idx % NR_;
    int n   = g * NCOL + nl2;
    float w = w_rec[(size_t)k * NR_ + n];
    if (k == n) w = 0.0f;
    wrec_t[idx] = __builtin_bit_cast(__bf16, f2bf_bits(w));
  }
  for (int idx = tid; idx < NCOL * NI_; idx += 256) {
    int nl2 = idx / NI_, k = idx % NI_;
    win_t[idx] = __builtin_bit_cast(__bf16, f2bf_bits(w_in[(size_t)k * NR_ + g * NCOL + nl2]));
  }
  __syncthreads();

  const int wave  = tid >> 5, lane = tid & 31;
  const int mt    = wave >> 2, nt = wave & 3;     // 2 M-tiles x 4 N-tiles
  const int row16 = lane & 15, half = lane >> 4;

  const int bA = mt * 16 + row16;                 // A-fragment batch row
  const int nl = nt * 16 + row16;                 // B/D local column
  const int ng = g * NCOL + nl;                   // global neuron index

  v8f vst = {};   // membrane potential, WMMA D layout
  v8f zst = {};   // previous spike, WMMA D layout

  for (int t = 0; t < T_; ++t) {
    const __bf16* zcur = (const __bf16*)zbuf + (size_t)(t & 1) * (B_ * NR_);
    unsigned short* znxt = zbuf + (size_t)((t + 1) & 1) * (B_ * NR_);

    v8f acc = {};

    // Feedforward: zin[t] @ w_in  (K = 128, 4 k-steps, pre-converted bf16)
    const __bf16* arow_in = zin + ((size_t)t * B_ + bA) * NI_;
    const __bf16* brow_in = win_t + (size_t)nl * NI_;
    #pragma unroll
    for (int kk = 0; kk < NI_ / 32; ++kk) {
      const int K0 = kk * 32 + half * 8;
      const int K1 = K0 + 16;
      v8bf alo = *(const v8bf*)(arow_in + K0);
      v8bf ahi = *(const v8bf*)(arow_in + K1);
      v8bf blo = *(const v8bf*)(brow_in + K0);
      v8bf bhi = *(const v8bf*)(brow_in + K1);
      v16bf a = __builtin_shufflevector(alo, ahi, 0,1,2,3,4,5,6,7,8,9,10,11,12,13,14,15);
      v16bf b = __builtin_shufflevector(blo, bhi, 0,1,2,3,4,5,6,7,8,9,10,11,12,13,14,15);
      acc = __builtin_amdgcn_wmma_f32_16x16x32_bf16(false, a, false, b,
                                                    (short)0, acc, false, false);
    }

    // Recurrent: z @ w_rec  (K = 512, 16 k-steps; B from LDS)
    const __bf16* zrow = zcur + (size_t)bA * NR_;
    const __bf16* brow = wrec_t + (size_t)nl * NR_;
    #pragma unroll 4
    for (int kk = 0; kk < NR_ / 32; ++kk) {
      const int K0 = kk * 32 + half * 8;
      const int K1 = K0 + 16;
      v8bf alo = *(const v8bf*)(zrow + K0);
      v8bf ahi = *(const v8bf*)(zrow + K1);
      v8bf blo = *(const v8bf*)(brow + K0);
      v8bf bhi = *(const v8bf*)(brow + K1);
      v16bf a = __builtin_shufflevector(alo, ahi, 0,1,2,3,4,5,6,7,8,9,10,11,12,13,14,15);
      v16bf b = __builtin_shufflevector(blo, bhi, 0,1,2,3,4,5,6,7,8,9,10,11,12,13,14,15);
      acc = __builtin_amdgcn_wmma_f32_16x16x32_bf16(false, a, false, b,
                                                    (short)0, acc, false, false);
    }

    // LIF update in registers (D layout: VGPR r -> M = r + 8*half)
    #pragma unroll
    for (int r = 0; r < 8; ++r) {
      const int brow_d = mt * 16 + half * 8 + r;
      float v = DECAY_V * vst[r] + acc[r] - zst[r];   // THR = 1
      bool  sp = (v > 1.0f);                          // spike((v-THR)/THR)
      float z = sp ? 1.0f : 0.0f;
      vst[r] = v;
      zst[r] = z;
      const size_t o = ((size_t)brow_d * T_ + t) * NR_ + ng;
      voltages[o] = v;
      spikes[o]   = z;
      znxt[(size_t)brow_d * NR_ + ng] = sp ? (unsigned short)0x3F80u
                                           : (unsigned short)0u;   // bf16 1.0 / 0.0
    }

    // Grid barrier (8 co-resident WGs): fence, count arrivals, flip generation.
    __threadfence();
    __syncthreads();
    if (tid == 0) {
      unsigned gen = __hip_atomic_load(&bar[1], __ATOMIC_ACQUIRE, __HIP_MEMORY_SCOPE_AGENT);
      unsigned arrived = __hip_atomic_fetch_add(&bar[0], 1u, __ATOMIC_ACQ_REL,
                                                __HIP_MEMORY_SCOPE_AGENT);
      if (arrived == NWG - 1) {
        __hip_atomic_store(&bar[0], 0u, __ATOMIC_RELAXED, __HIP_MEMORY_SCOPE_AGENT);
        __hip_atomic_fetch_add(&bar[1], 1u, __ATOMIC_RELEASE, __HIP_MEMORY_SCOPE_AGENT);
      } else {
        while (__hip_atomic_load(&bar[1], __ATOMIC_ACQUIRE,
                                 __HIP_MEMORY_SCOPE_AGENT) == gen)
          __builtin_amdgcn_s_sleep(2);
      }
    }
    __syncthreads();
  }
}

// ---------------------------------------------------------------------------
// Readout: proj = spikes @ w_out + b_out, then sequential exp filter per batch.
// One block per batch, one thread per neuron.
// ---------------------------------------------------------------------------
__global__ void __launch_bounds__(512) lsnn_readout_kernel(
    const float* __restrict__ spikes,   // [B][T][NR]
    const float* __restrict__ w_out,    // [NR]
    const float* __restrict__ b_out,    // [1]
    float* __restrict__ pred)           // [B][T]
{
  __shared__ float sdata[512];
  const int b = blockIdx.x, tid = threadIdx.x;
  const float wo = w_out[tid];
  const float b0 = b_out[0];
  float a = 0.0f;
  for (int t = 0; t < T_; ++t) {
    float z = spikes[((size_t)b * T_ + t) * NR_ + tid];
    sdata[tid] = z * wo;
    __syncthreads();
    for (int s = 256; s > 0; s >>= 1) {
      if (tid < s) sdata[tid] += sdata[tid + s];
      __syncthreads();
    }
    if (tid == 0) {
      a = OUT_DECAY * a + (1.0f - OUT_DECAY) * (sdata[0] + b0);
      pred[(size_t)b * T_ + t] = a;
    }
    __syncthreads();
  }
}

extern "C" void kernel_launch(void* const* d_in, const int* in_sizes, int n_in,
                              void* d_out, int out_size, void* d_ws, size_t ws_size,
                              hipStream_t stream) {
  (void)in_sizes; (void)n_in; (void)out_size; (void)ws_size;
  const float* inputs = (const float*)d_in[0];
  const float* w_in   = (const float*)d_in[1];
  const float* w_rec  = (const float*)d_in[2];
  const float* w_out  = (const float*)d_in[3];
  const float* b_out  = (const float*)d_in[4];

  float* out      = (float*)d_out;
  float* voltages = out;
  float* spikes   = out + (size_t)B_ * T_ * NR_;
  float* pred     = out + (size_t)2 * B_ * T_ * NR_;

  // Workspace layout: zin (8.192 MB) | zbuf (64 KB) | barrier (8 B)
  unsigned short* zin  = (unsigned short*)d_ws;
  unsigned short* zbuf = (unsigned short*)((char*)d_ws + (size_t)T_ * B_ * NI_ * 2);
  unsigned*       bar  = (unsigned*)((char*)zbuf + (size_t)2 * B_ * NR_ * 2);

  lsnn_init<<<1, 256, 0, stream>>>((unsigned*)zbuf, bar);

  const size_t nconv = (size_t)T_ * B_ * NI_;
  lsnn_convert_inputs<<<(unsigned)((nconv + 255) / 256), 256, 0, stream>>>(inputs, zin);

  const size_t smem = (size_t)(NCOL * NR_ + NCOL * NI_) * sizeof(__bf16); // 80 KB
  lsnn_step_kernel<<<NWG, 256, smem, stream>>>((const __bf16*)zin, w_in, w_rec, zbuf,
                                               bar, voltages, spikes);

  lsnn_readout_kernel<<<B_, 512, 0, stream>>>(spikes, w_out, b_out, pred);
}
